// Seq2SeqCritic_88046829568820
// MI455X (gfx1250) — matrix operand order repaired and checked by reading, not compiled, not run on hardware
//
#include <hip/hip_runtime.h>
#include <hip/hip_bf16.h>
#include <math.h>

// ---------------------------------------------------------------------------
// Types for CDNA5 WMMA (wave32):  D(f32 16x16) = A(bf16 16x32) x B(bf16 32x16) + C
// ---------------------------------------------------------------------------
typedef __attribute__((ext_vector_type(16))) __bf16 v16bf;
typedef __attribute__((ext_vector_type(8)))  float  v8f;

__device__ __forceinline__ unsigned short bf16bits(float f) {
    union { float f; unsigned u; } a; a.f = f;
    unsigned r = a.u + 0x7fffu + ((a.u >> 16) & 1u);   // round-to-nearest-even
    return (unsigned short)(r >> 16);
}
__device__ __forceinline__ __bf16 f2bf(float f) {
    return __builtin_bit_cast(__bf16, bf16bits(f));
}
__device__ __forceinline__ unsigned pack2bf(float lo, float hi) {
    return (unsigned)bf16bits(lo) | ((unsigned)bf16bits(hi) << 16);
}

// async copy 16B global -> LDS (CDNA5 Tensor/Async path, tracked on ASYNCcnt)
__device__ __forceinline__ void async_g2l_b128(void* lds_dst, const void* gsrc) {
    unsigned ldsoff = (unsigned)(size_t)lds_dst;           // addrspace(3) offset
    unsigned long long ga = (unsigned long long)(size_t)gsrc;
    asm volatile("global_load_async_to_lds_b128 %0, %1, off"
                 :: "v"(ldsoff), "v"(ga) : "memory");
}
__device__ __forceinline__ void wait_async0() {
    asm volatile("s_wait_asynccnt 0" ::: "memory");
}

// ---------------------------------------------------------------------------
// Weight prep: W[K,N] f32 (row-major) -> Wt[N,K] bf16 (row-major, K-contiguous)
// 32x32 LDS transpose, block (32,8). K,N multiples of 32.
// ---------------------------------------------------------------------------
__global__ __launch_bounds__(256)
void transpose_bf16_kernel(const float* __restrict__ W, __bf16* __restrict__ Wt,
                           int K, int N)
{
    __shared__ float tile[32][33];
    const int kb = blockIdx.y * 32, nb = blockIdx.x * 32;
#pragma unroll
    for (int i = threadIdx.y; i < 32; i += 8)
        tile[i][threadIdx.x] = W[(size_t)(kb + i) * N + nb + threadIdx.x];
    __syncthreads();
#pragma unroll
    for (int i = threadIdx.y; i < 32; i += 8)
        Wt[(size_t)(nb + i) * K + kb + threadIdx.x] = f2bf(tile[threadIdx.x][i]);
}

// ---------------------------------------------------------------------------
// WMMA GEMM: Cout[M,N] = A[M,K](f32,row-major,lda) * Bt[N,K](bf16, pre-transposed,
// row stride ldbt) (+ Cin) (+ bias[N]).  REQUIRES: M % BM == 0, N % BN == 0,
// K % 32 == 0, pointers 16B aligned, lda mult of 4, ldbt mult of 8.
// Double-buffered LDS; B tile staged with async global->LDS b128 copies.
// ---------------------------------------------------------------------------
template<int BM, int BN, int WAVES_M, int WAVES_N>
__global__ __launch_bounds__(WAVES_M * WAVES_N * 32)
void gemm_wmma_kernel(const float* __restrict__ A, int lda,
                      const __bf16* __restrict__ Bt, int ldbt,
                      const float* __restrict__ Cin,
                      const float* __restrict__ bias,
                      float* __restrict__ Cout, int ldc,
                      int K)
{
    constexpr int TK  = 32;
    constexpr int TKP = 40;                       // row stride 80B (16B-aligned rows)
    constexpr int NT  = WAVES_M * WAVES_N * 32;
    constexpr int WM  = BM / WAVES_M;
    constexpr int WN  = BN / WAVES_N;
    constexpr int MT  = WM / 16;
    constexpr int NTL = WN / 16;
    constexpr int ACH = (BM * 8) / NT;            // A-tile float4 chunks per thread

    __shared__ alignas(16) __bf16 As[2][BM * TKP];   // [BM][TK] K-contiguous
    __shared__ alignas(16) __bf16 Bs[2][BN * TKP];   // [BN][TK] K-contiguous

    const int tid  = threadIdx.x;
    const int lane = tid & 31;
    const int wid  = tid >> 5;
    const int wm   = (wid / WAVES_N) * WM;
    const int wn   = (wid % WAVES_N) * WN;
    const int m0   = blockIdx.y * BM;
    const int n0   = blockIdx.x * BN;

    // A stage: all global loads issued first (latency overlap), then pack+store
    auto stageA = [&](int k0, int buf) {
        float4 vv[ACH];
#pragma unroll
        for (int c2 = 0; c2 < ACH; ++c2) {
            int idx = tid + c2 * NT;
            int r = idx >> 3, c = (idx & 7) << 2;
            vv[c2] = *(const float4*)(A + (size_t)(m0 + r) * lda + k0 + c);
        }
#pragma unroll
        for (int c2 = 0; c2 < ACH; ++c2) {
            int idx = tid + c2 * NT;
            int r = idx >> 3, c = (idx & 7) << 2;
            uint2 pp; pp.x = pack2bf(vv[c2].x, vv[c2].y); pp.y = pack2bf(vv[c2].z, vv[c2].w);
            *(uint2*)&As[buf][r * TKP + c] = pp;
        }
    };
    auto stageB = [&](int k0, int buf) {
#pragma unroll
        for (int idx = tid; idx < BN * 4; idx += NT) {
            int n = idx >> 2, c = (idx & 3) << 3;
            async_g2l_b128(&Bs[buf][n * TKP + c],
                           Bt + (size_t)(n0 + n) * ldbt + k0 + c);
        }
    };

    v8f acc[MT][NTL];
    const v8f vz = {0.f,0.f,0.f,0.f,0.f,0.f,0.f,0.f};
#pragma unroll
    for (int i = 0; i < MT; ++i)
#pragma unroll
        for (int j = 0; j < NTL; ++j) acc[i][j] = vz;

    const int ntiles = K >> 5;
    stageA(0, 0);
    stageB(0, 0);

    for (int kt = 0; kt < ntiles; ++kt) {
        const int cur = kt & 1;
        wait_async0();                 // tile kt's async B copies done (this wave)
        __syncthreads();               // all waves' tile kt staging visible
        if (kt + 1 < ntiles) {         // prefetch tile kt+1 into other buffer
            stageA((kt + 1) << 5, cur ^ 1);
            stageB((kt + 1) << 5, cur ^ 1);
        }

        union Frag { v16bf v; unsigned u[8]; };
        // A frags: lane L<16 -> row M=L, K {0..7,16..23}; L>=16 -> K {8..15,24..31}
        Frag a[MT];
        const int arow = wm + (lane & 15);
        const int akb  = (lane >> 4) << 3;
#pragma unroll
        for (int mt = 0; mt < MT; ++mt) {
#pragma unroll
            for (int i = 0; i < 8; ++i) {
                int kk = akb + ((i >> 2) << 4) + ((i & 3) << 1);
                a[mt].u[i] = *(const unsigned*)&As[cur][(arow + mt * 16) * TKP + kk];
            }
        }
        // B frags: lane L<16 -> col N=L, K {0..7,16..23}; L>=16 -> K {8..15,24..31}
        const int bcol = wn + (lane & 15);
        const int bkb  = (lane >> 4) << 3;
#pragma unroll
        for (int nt = 0; nt < NTL; ++nt) {
            Frag bf;
#pragma unroll
            for (int i = 0; i < 8; ++i)
                bf.u[i] = *(const unsigned*)&Bs[cur][(bcol + nt * 16) * TKP + bkb + ((i >> 2) << 4) + ((i & 3) << 1)];
#pragma unroll
            for (int mt = 0; mt < MT; ++mt)
                acc[mt][nt] = __builtin_amdgcn_wmma_f32_16x16x32_bf16(
                    false, a[mt].v, false, bf.v, (short)0, acc[mt][nt], false, false);
        }
    }

    // epilogue: lanes 0-15: N=lane, M=vgpr; lanes 16-31: N=lane-16, M=vgpr+8
#pragma unroll
    for (int mt = 0; mt < MT; ++mt) {
#pragma unroll
        for (int nt = 0; nt < NTL; ++nt) {
            int n  = n0 + wn + nt * 16 + (lane & 15);
            int mb = m0 + wm + mt * 16 + ((lane >> 4) << 3);
            float bv = bias ? bias[n] : 0.f;
#pragma unroll
            for (int i = 0; i < 8; ++i) {
                int m = mb + i;
                float v = acc[mt][nt][i] + bv;
                if (Cin) v += Cin[(size_t)m * ldc + n];
                Cout[(size_t)m * ldc + n] = v;
            }
        }
    }
}

// ---------------------------------------------------------------------------
// LSTM gate nonlinearity: g[B,4H] (i,j,f,o chunks of H), c in-place, h -> hout
// ---------------------------------------------------------------------------
__global__ void lstm_gate_kernel(const float* __restrict__ G,
                                 float* __restrict__ c,
                                 float* __restrict__ hout, int ldh)
{
    int idx = blockIdx.x * blockDim.x + threadIdx.x;   // B*H = 32768 threads
    int b = idx >> 10, hh = idx & 1023;
    const float* g = G + (size_t)b * 4096;
    float gi = g[hh], gj = g[1024 + hh], gf = g[2048 + hh], go = g[3072 + hh];
    float si = 1.f / (1.f + __expf(-gi));
    float sf = 1.f / (1.f + __expf(-(gf + 1.f)));      // forget_bias = 1.0
    float so = 1.f / (1.f + __expf(-go));
    float cn = sf * c[idx] + si * tanhf(gj);
    c[idx] = cn;
    hout[(size_t)b * ldh + hh] = so * tanhf(cn);
}

// scores[b,te] = dot(keys_tm[te,b,:], h[b,:])  (keys time-major [Te,B,H], h stride 2048)
__global__ void attn_scores_kernel(const float* __restrict__ keys_tm,
                                   const float* __restrict__ h,
                                   float* __restrict__ scores)
{
    const int b = blockIdx.x;
    const int lane = threadIdx.x & 31, wave = threadIdx.x >> 5;
    const int te = blockIdx.y * 8 + wave;
    const float* kp = keys_tm + ((size_t)te * 32 + b) * 1024;
    const float* hp = h + (size_t)b * 2048;
    float s = 0.f;
    for (int k = lane; k < 1024; k += 32) s += kp[k] * hp[k];
#pragma unroll
    for (int o = 16; o; o >>= 1) s += __shfl_xor(s, o, 32);
    if (lane == 0) scores[b * 256 + te] = s;
}

// softmax over 256 scores + ctx[b,h] = sum_te alpha * values_tm[te,b,h]; ctx -> HC[:,1024:]
__global__ void attn_ctx_kernel(const float* __restrict__ scores,
                                const float* __restrict__ vals_tm,
                                float* __restrict__ HC)
{
    const int b = blockIdx.x, tid = threadIdx.x;
    const int lane = tid & 31, wave = tid >> 5;
    __shared__ float sa[256];
    __shared__ float red[8];
    float v = scores[b * 256 + tid];
    float m = v;
#pragma unroll
    for (int o = 16; o; o >>= 1) m = fmaxf(m, __shfl_xor(m, o, 32));
    if (lane == 0) red[wave] = m;
    __syncthreads();
    float bm = red[0];
#pragma unroll
    for (int w = 1; w < 8; ++w) bm = fmaxf(bm, red[w]);
    float e = __expf(v - bm);
    __syncthreads();
    float s = e;
#pragma unroll
    for (int o = 16; o; o >>= 1) s += __shfl_xor(s, o, 32);
    if (lane == 0) red[wave] = s;
    __syncthreads();
    float ssum = 0.f;
#pragma unroll
    for (int w = 0; w < 8; ++w) ssum += red[w];
    sa[tid] = e / ssum;
    __syncthreads();
    for (int hh = tid; hh < 1024; hh += 256) {
        float a = 0.f;
        const float* vp = vals_tm + (size_t)b * 1024 + hh;
        for (int te = 0; te < 256; ++te) a += sa[te] * vp[(size_t)te * 32 * 1024];
        HC[(size_t)b * 2048 + 1024 + hh] = a;
    }
}

__global__ void zero_kernel(float* __restrict__ p, int n) {
    int i = blockIdx.x * blockDim.x + threadIdx.x;
    if (i < n) p[i] = 0.f;
}

// copy encoder final h [32,1024] into HC[:, :1024] (stride 2048)
__global__ void init_dec_h_kernel(const float* __restrict__ h2last, float* __restrict__ HC) {
    int idx = blockIdx.x * blockDim.x + threadIdx.x;
    int b = idx >> 10, hh = idx & 1023;
    HC[(size_t)b * 2048 + hh] = h2last[idx];
}

// ---------------------------------------------------------------------------
// Host-side wrappers
// ---------------------------------------------------------------------------
static inline void gemm_big(const float* A, int lda, const __bf16* Bt, int ldbt,
                            const float* Cin, const float* bias,
                            float* C, int ldc, int M, int N, int K, hipStream_t s)
{
    dim3 g(N / 128, M / 128);
    gemm_wmma_kernel<128, 128, 4, 2><<<g, 256, 0, s>>>(A, lda, Bt, ldbt, Cin, bias, C, ldc, K);
}

static inline void gemm_thin(const float* A, int lda, const __bf16* Bt, int ldbt,
                             const float* Cin, const float* bias,
                             float* C, int ldc, int M, int N, int K, hipStream_t s)
{
    dim3 g(N / 128, M / 32);
    gemm_wmma_kernel<32, 128, 1, 4><<<g, 128, 0, s>>>(A, lda, Bt, ldbt, Cin, bias, C, ldc, K);
}

extern "C" void kernel_launch(void* const* d_in, const int* in_sizes, int n_in,
                              void* d_out, int out_size, void* d_ws, size_t ws_size,
                              hipStream_t stream)
{
    (void)in_sizes; (void)n_in; (void)out_size; (void)ws_size;
    const float* enc_in = (const float*)d_in[0];   // [256,32,512]
    const float* dec_in = (const float*)d_in[1];   // [128,32,512]
    const float* Wx0    = (const float*)d_in[2];   // [512,4096]
    const float* Wh0    = (const float*)d_in[3];   // [1024,4096]
    const float* b0     = (const float*)d_in[4];
    const float* Wx1    = (const float*)d_in[5];   // [1024,4096]
    const float* Wh1    = (const float*)d_in[6];
    const float* b1     = (const float*)d_in[7];
    const float* dWx    = (const float*)d_in[8];   // [1536,4096]
    const float* dWh    = (const float*)d_in[9];   // [1024,4096]
    const float* db     = (const float*)d_in[10];
    const float* Wk     = (const float*)d_in[11];  // [1024,1024]
    const float* Wa     = (const float*)d_in[12];  // [2048,1024]
    const float* ba     = (const float*)d_in[13];
    const float* Wout   = (const float*)d_in[14];  // [1024,8192]
    const float* bout   = (const float*)d_in[15];
    float* out = (float*)d_out;                    // [128,32,8192]

    // ---------------- workspace carve-up ----------------
    float* ws   = (float*)d_ws;
    float* XG   = ws; ws += (size_t)8192 * 4096;   // Xg0 -> Xg1 -> Xd (reused)
    float* h1   = ws; ws += (size_t)8192 * 1024;   // layer0 h sequence [Te,B,H]
    float* h2   = ws; ws += (size_t)8192 * 1024;   // layer1 h sequence (values, time-major)
    float* keys = ws; ws += (size_t)8192 * 1024;   // keys time-major [Te,B,H]
    float* Att  = ws; ws += (size_t)4096 * 1024;   // decoder attn outputs [Td,B,H]
    float* c0   = ws; ws += 32 * 1024;             // layer0 cell state
    float* cS   = ws; ws += 32 * 1024;             // layer1 cell -> decoder cell
    float* HC   = ws; ws += 32 * 2048;             // [h | ctx] concat buffer
    float* Gb   = ws; ws += 32 * 4096;             // per-step gate buffer
    float* zb   = ws; ws += 32 * 1024;             // zero attn for t=0
    float* sc   = ws; ws += 32 * 256;              // attention scores

    uintptr_t p = ((uintptr_t)ws + 15) & ~(uintptr_t)15;
    __bf16* wsb  = (__bf16*)p;                     // bf16 transposed weights [N,K]
    __bf16* Wx0t = wsb; wsb += (size_t)4096 * 512;
    __bf16* Wh0t = wsb; wsb += (size_t)4096 * 1024;
    __bf16* Wx1t = wsb; wsb += (size_t)4096 * 1024;
    __bf16* Wh1t = wsb; wsb += (size_t)4096 * 1024;
    __bf16* dWxt = wsb; wsb += (size_t)4096 * 1536;
    __bf16* dWht = wsb; wsb += (size_t)4096 * 1024;
    __bf16* Wkt  = wsb; wsb += (size_t)1024 * 1024;
    __bf16* Wat  = wsb; wsb += (size_t)1024 * 2048;
    __bf16* Woutt= wsb; wsb += (size_t)8192 * 1024;

    // ---------------- weight prep (bf16 + transpose) ----------------
    {
        dim3 b(32, 8);
        transpose_bf16_kernel<<<dim3(4096/32,  512/32), b, 0, stream>>>(Wx0,  Wx0t,  512, 4096);
        transpose_bf16_kernel<<<dim3(4096/32, 1024/32), b, 0, stream>>>(Wh0,  Wh0t, 1024, 4096);
        transpose_bf16_kernel<<<dim3(4096/32, 1024/32), b, 0, stream>>>(Wx1,  Wx1t, 1024, 4096);
        transpose_bf16_kernel<<<dim3(4096/32, 1024/32), b, 0, stream>>>(Wh1,  Wh1t, 1024, 4096);
        transpose_bf16_kernel<<<dim3(4096/32, 1536/32), b, 0, stream>>>(dWx,  dWxt, 1536, 4096);
        transpose_bf16_kernel<<<dim3(4096/32, 1024/32), b, 0, stream>>>(dWh,  dWht, 1024, 4096);
        transpose_bf16_kernel<<<dim3(1024/32, 1024/32), b, 0, stream>>>(Wk,   Wkt,  1024, 1024);
        transpose_bf16_kernel<<<dim3(1024/32, 2048/32), b, 0, stream>>>(Wa,   Wat,  2048, 1024);
        transpose_bf16_kernel<<<dim3(8192/32, 1024/32), b, 0, stream>>>(Wout, Woutt,1024, 8192);
    }

    const int BH = 32 * 1024;
    zero_kernel<<<BH / 256, 256, 0, stream>>>(c0, BH);
    zero_kernel<<<BH / 256, 256, 0, stream>>>(cS, BH);
    zero_kernel<<<BH / 256, 256, 0, stream>>>(zb, BH);

    // ---------------- encoder layer 0 ----------------
    gemm_big(enc_in, 512, Wx0t, 512, nullptr, b0, XG, 4096, 8192, 4096, 512, stream);
    for (int t = 0; t < 256; ++t) {
        const float* Gt;
        if (t == 0) {
            Gt = XG;                                   // h_{-1}=0 -> gates = Xg_0
        } else {
            gemm_thin(h1 + (size_t)(t - 1) * BH, 1024, Wh0t, 1024,
                      XG + (size_t)t * 32 * 4096, nullptr, Gb, 4096, 32, 4096, 1024, stream);
            Gt = Gb;
        }
        lstm_gate_kernel<<<BH / 256, 256, 0, stream>>>(Gt, c0, h1 + (size_t)t * BH, 1024);
    }

    // ---------------- encoder layer 1 ----------------
    gemm_big(h1, 1024, Wx1t, 1024, nullptr, b1, XG, 4096, 8192, 4096, 1024, stream);
    for (int t = 0; t < 256; ++t) {
        const float* Gt;
        if (t == 0) {
            Gt = XG;
        } else {
            gemm_thin(h2 + (size_t)(t - 1) * BH, 1024, Wh1t, 1024,
                      XG + (size_t)t * 32 * 4096, nullptr, Gb, 4096, 32, 4096, 1024, stream);
            Gt = Gb;
        }
        lstm_gate_kernel<<<BH / 256, 256, 0, stream>>>(Gt, cS, h2 + (size_t)t * BH, 1024);
    }

    // keys = values @ Wk, computed time-major: [Te*B,H] x [H,H]
    gemm_big(h2, 1024, Wkt, 1024, nullptr, nullptr, keys, 1024, 8192, 1024, 1024, stream);

    // decoder x-half of dec_Wx precomputed for all steps (+ bias)
    // x part = original rows 0..511  -> transposed: k offset 0, row stride 1536
    gemm_big(dec_in, 512, dWxt, 1536, nullptr, db, XG, 4096, 4096, 4096, 512, stream);

    // decoder initial h = encoder layer1 final h; c continues in cS
    init_dec_h_kernel<<<BH / 256, 256, 0, stream>>>(h2 + (size_t)255 * BH, HC);

    // ---------------- attention decoder ----------------
    const __bf16* dWxt_a = dWxt + 512;             // attn part: original rows 512..1535
    for (int t = 0; t < 128; ++t) {
        const float* aprev = (t == 0) ? zb : (Att + (size_t)(t - 1) * BH);
        // gates = Xd_t + attn_prev @ Wx_a + h @ dec_Wh
        gemm_thin(aprev, 1024, dWxt_a, 1536, XG + (size_t)t * 32 * 4096, nullptr,
                  Gb, 4096, 32, 4096, 1024, stream);
        gemm_thin(HC, 2048, dWht, 1024, Gb, nullptr, Gb, 4096, 32, 4096, 1024, stream);
        lstm_gate_kernel<<<BH / 256, 256, 0, stream>>>(Gb, cS, HC, 2048);   // h -> HC[:, :1024]
        attn_scores_kernel<<<dim3(32, 32), 256, 0, stream>>>(keys, HC, sc);
        attn_ctx_kernel<<<32, 256, 0, stream>>>(sc, h2, HC);                // ctx -> HC[:,1024:]
        // attn = [h|ctx] @ Wa + ba
        gemm_thin(HC, 2048, Wat, 2048, nullptr, ba, Att + (size_t)t * BH, 1024,
                  32, 1024, 2048, stream);
    }

    // logits = outs @ Wout + bout   ([4096,1024] x [1024,8192])
    gemm_big(Att, 1024, Woutt, 1024, nullptr, bout, out, 8192, 4096, 8192, 1024, stream);
}
